// RGCNBlock_41446434406671
// MI455X (gfx1250) — matrix operand order repaired and checked by reading, not compiled, not run on hardware
//
#include <hip/hip_runtime.h>
#include <hip/hip_bf16.h>

// ---------------------------------------------------------------------------
// RGCN block for MI455X (gfx1250, wave32).
// - All GEMMs on v_wmma_f32_16x16x32_bf16 (f32 accumulate).
// - x converted to bf16 once -> 51 MB gather table is L2-resident (192 MB L2).
// - Message stage: 64 edges/block; W_r staged into LDS by the Tensor Data
//   Mover with DOUBLE BUFFERING: two 32 KB LDS buffers, quarter-K tiles,
//   s_wait_tensorcnt(1) overlaps the next DMA with current WMMA work.
// - Segment-sum via global_atomic_add_f32.
// ---------------------------------------------------------------------------

typedef __bf16 bf16;
typedef __attribute__((ext_vector_type(16))) __bf16 bf16x16;
typedef __attribute__((ext_vector_type(8)))  __bf16 bf16x8;
typedef __attribute__((ext_vector_type(8)))  float  f32x8;
typedef __attribute__((ext_vector_type(4)))  unsigned int u32x4;
typedef __attribute__((ext_vector_type(8)))  int i32x8;
typedef __attribute__((ext_vector_type(4)))  int i32x4;

static constexpr int D  = 256;    // hidden dim
static constexpr int FF = 1024;   // ffn inner dim
static constexpr int KQ = 64;     // bf16 K values staged per TDM quarter

// ---- WMMA helpers ---------------------------------------------------------

__device__ __forceinline__ f32x8 wmma_bf16(bf16x16 a, bf16x16 b, f32x8 c) {
  // 8 args: (neg_a, A, neg_b, B, c_mod, C, reuse_a, reuse_b)
  return __builtin_amdgcn_wmma_f32_16x16x32_bf16(
      false, a, false, b, (short)0, c, false, false);
}

// A-matrix 16x32 bf16 fragment (ISA 7.12.2):
// lane(col,khalf): row M = lane&15; elements 0..7  = K kbase+8*khalf..+7,
//                                   elements 8..15 = K kbase+16+8*khalf..+7
__device__ __forceinline__ bf16x16 load_frag_a(const bf16* row, int kbase, int khalf) {
  bf16x8 lo = *reinterpret_cast<const bf16x8*>(row + kbase + khalf * 8);
  bf16x8 hi = *reinterpret_cast<const bf16x8*>(row + kbase + 16 + khalf * 8);
  bf16x16 a;
#pragma unroll
  for (int i = 0; i < 8; ++i) { a[i] = lo[i]; a[i + 8] = hi[i]; }
  return a;
}

// B-matrix 32x16 bf16 fragment from an output-column-major weight row
// (colrow = &Wt[n*K]): lane holds column n = lane&15; elements 0..7 = K
// kbase+8*khalf..+7, elements 8..15 = K kbase+16+8*khalf..+7. Works for global or LDS-backed pointers.
__device__ __forceinline__ bf16x16 load_frag_b(const bf16* colrow, int kbase, int khalf) {
  bf16x8 lo = *reinterpret_cast<const bf16x8*>(colrow + kbase + khalf * 8);
  bf16x8 hi = *reinterpret_cast<const bf16x8*>(colrow + kbase + 16 + khalf * 8);
  bf16x16 b;
#pragma unroll
  for (int i = 0; i < 8; ++i) { b[i] = lo[i]; b[i + 8] = hi[i]; }
  return b;
}

// ---- Tensor Data Mover: 2-D tile Global -> LDS ----------------------------
// D# packing per CDNA5 ISA:
//   group0: [1:0]=count=1, [63:32]=lds_addr, [120:64]=global_addr, [127:126]=2
//   group1: [17:16]=data_size(2 -> 4B), [79:48]=tensor_dim0,
//           [111:80]=tensor_dim1, [127:112]=tile_dim0, [143:128]=tile_dim1,
//           [207:160]=tensor_dim0_stride
__device__ __forceinline__ void tdm_stage_2d(unsigned lds_off, const void* gptr,
                                             unsigned tile_d0_dw, unsigned tile_d1,
                                             unsigned stride_dw) {
  unsigned long long ga = (unsigned long long)(uintptr_t)gptr;
  u32x4 g0;
  g0[0] = 1u;                                                  // count=1 (valid)
  g0[1] = lds_off;                                             // lds_addr
  g0[2] = (unsigned)(ga & 0xffffffffull);                      // global_addr lo
  g0[3] = (unsigned)((ga >> 32) & 0x01ffffffull) | (2u << 30); // addr hi + type=2
  i32x8 g1;
  g1[0] = (int)(2u << 16);                                     // data_size = 4B
  g1[1] = (int)((tile_d0_dw & 0xffffu) << 16);                 // tensor_dim0 lo
  g1[2] = (int)(((tile_d0_dw >> 16) & 0xffffu) |
                ((tile_d1 & 0xffffu) << 16));                  // dim0 hi | dim1 lo
  g1[3] = (int)(((tile_d1 >> 16) & 0xffffu) |
                ((tile_d0_dw & 0xffffu) << 16));               // dim1 hi | tile_dim0
  g1[4] = (int)(tile_d1 & 0xffffu);                            // tile_dim1 (dim2=0)
  g1[5] = (int)stride_dw;                                      // dim0_stride lo
  g1[6] = 0;
  g1[7] = 0;
  i32x4 z4 = {0, 0, 0, 0};
#if defined(__clang_major__) && (__clang_major__ >= 23)
  i32x8 z8 = {0, 0, 0, 0, 0, 0, 0, 0};
  __builtin_amdgcn_tensor_load_to_lds(g0, g1, z4, z4, z8, 0);
#else
  __builtin_amdgcn_tensor_load_to_lds(g0, g1, z4, z4, 0);
#endif
}

// ---- conversion kernels ---------------------------------------------------

__global__ __launch_bounds__(256) void cvt_bf16_kernel(const float* __restrict__ in,
                                                       bf16* __restrict__ out, size_t n) {
  size_t i = (size_t)blockIdx.x * blockDim.x + threadIdx.x;
  if (i < n) out[i] = (bf16)in[i];
}

// relation weights are used as x @ W (not W^T): Wt[r][j][k] = W[r][k][j]
__global__ __launch_bounds__(256) void cvt_rel_kernel(const float* __restrict__ w,
                                                      bf16* __restrict__ wt, int R) {
  size_t idx = (size_t)blockIdx.x * blockDim.x + threadIdx.x;
  size_t total = (size_t)R * D * D;
  if (idx >= total) return;
  int r = (int)(idx / ((size_t)D * D));
  int rem = (int)(idx % ((size_t)D * D));
  int j = rem / D, k = rem % D;
  wt[idx] = (bf16)w[(size_t)r * D * D + (size_t)k * D + j];
}

// ---- stage 1: agg = x @ self_loop_w^T + x  (WMMA) -------------------------

__global__ __launch_bounds__(256) void selfloop_kernel(const float* __restrict__ x,
                                                       const bf16* __restrict__ xb,
                                                       const bf16* __restrict__ wself,
                                                       float* __restrict__ agg, int N) {
  const int tile = blockIdx.x;            // 16 node rows
  const int wave = threadIdx.x >> 5;
  const int lane = threadIdx.x & 31;
  const int col  = lane & 15;
  const int khalf = lane >> 4;

  int arow_node = tile * 16 + col;
  if (arow_node >= N) arow_node = N - 1;
  const bf16* arow = xb + (size_t)arow_node * D;

  const int nt0 = wave * 2;               // each wave: 2 of 16 column tiles
  const bf16* b0row = wself + (size_t)(nt0 * 16 + col) * D;
  const bf16* b1row = wself + (size_t)((nt0 + 1) * 16 + col) * D;

  f32x8 acc0 = {0, 0, 0, 0, 0, 0, 0, 0};
  f32x8 acc1 = {0, 0, 0, 0, 0, 0, 0, 0};
#pragma unroll
  for (int ks = 0; ks < D / 32; ++ks) {
    bf16x16 a = load_frag_a(arow, ks * 32, khalf);
    acc0 = wmma_bf16(a, load_frag_b(b0row, ks * 32, khalf), acc0);
    acc1 = wmma_bf16(a, load_frag_b(b1row, ks * 32, khalf), acc1);
  }

#pragma unroll
  for (int i = 0; i < 8; ++i) {
    int node = tile * 16 + i + 8 * khalf;
    if (node < N) {
      size_t o0 = (size_t)node * D + nt0 * 16 + col;
      size_t o1 = (size_t)node * D + (nt0 + 1) * 16 + col;
      agg[o0] = acc0[i] + x[o0];
      agg[o1] = acc1[i] + x[o1];
    }
  }
}

// ---- stage 2: per-relation messages + scatter-add -------------------------
// 64 edges per block (4 row-tiles). W_r is staged into LDS by the TDM in four
// quarter-K tiles (KQ=64 bf16 per column) with double buffering: two 32 KB
// buffers; s_wait_tensorcnt(1) releases the oldest in-flight DMA while the
// next one overlaps WMMA compute. Accumulators persist across stages.

__global__ __launch_bounds__(256) void msg_kernel(const bf16* __restrict__ xb,
                                                  const bf16* __restrict__ wrel,
                                                  const int* __restrict__ src,
                                                  const int* __restrict__ dst,
                                                  const float* __restrict__ norm,
                                                  float* __restrict__ agg,
                                                  int N, int E) {
  __shared__ bf16 wlds[2][KQ * D];        // 2 x 32 KB, col-major over K (stride KQ)

  const int tile = blockIdx.x;            // 64 edges
  const int r    = blockIdx.y;
  const int wave = threadIdx.x >> 5;
  const int lane = threadIdx.x & 31;
  const int col  = lane & 15;
  const int khalf = lane >> 4;
  const int base_e = tile * 64;

  // cache gathered A-row pointers for the 4 row-tiles
  const bf16* arow[4];
#pragma unroll
  for (int rt = 0; rt < 4; ++rt) {
    int e_row = base_e + rt * 16 + col;
    if (e_row >= E) e_row = E - 1;
    arow[rt] = xb + (size_t)src[(size_t)r * E + e_row] * D;
  }

  const bf16* wt = wrel + (size_t)r * D * D;
  const int nt0 = wave * 2;               // this wave's 2 column tiles

  f32x8 acc[4][2];
#pragma unroll
  for (int rt = 0; rt < 4; ++rt) {
    acc[rt][0] = (f32x8){0, 0, 0, 0, 0, 0, 0, 0};
    acc[rt][1] = (f32x8){0, 0, 0, 0, 0, 0, 0, 0};
  }

  // quarter-K tile: 32 dwords (=KQ bf16) x 256 columns, memory row stride
  // 128 dwords (full 256-bf16 column). Quarter q starts at byte q*KQ*2 in
  // each column. Pre-issue two DMAs, then rotate buffers.
  if (wave == 0) {
    tdm_stage_2d((unsigned)(uintptr_t)(void*)&wlds[0][0],
                 (const void*)((const char*)wt + 0 * KQ * 2),
                 /*tile_d0_dw=*/KQ / 2, /*tile_d1=*/D, /*stride_dw=*/D / 2);
    tdm_stage_2d((unsigned)(uintptr_t)(void*)&wlds[1][0],
                 (const void*)((const char*)wt + 1 * KQ * 2),
                 /*tile_d0_dw=*/KQ / 2, /*tile_d1=*/D, /*stride_dw=*/D / 2);
  }

#pragma unroll
  for (int q = 0; q < 4; ++q) {
    if (wave == 0) {
      if (q < 3) __builtin_amdgcn_s_wait_tensorcnt(1);  // oldest DMA done
      else       __builtin_amdgcn_s_wait_tensorcnt(0);  // last DMA done
    }
    __syncthreads();                      // buffer q&1 ready for all waves

    const bf16* bl0 = &wlds[q & 1][0] + (size_t)(nt0 * 16 + col) * KQ;
    const bf16* bl1 = &wlds[q & 1][0] + (size_t)((nt0 + 1) * 16 + col) * KQ;
#pragma unroll
    for (int rt = 0; rt < 4; ++rt) {
#pragma unroll
      for (int ks = 0; ks < KQ / 32; ++ks) {
        bf16x16 a = load_frag_a(arow[rt], q * KQ + ks * 32, khalf);
        acc[rt][0] = wmma_bf16(a, load_frag_b(bl0, ks * 32, khalf), acc[rt][0]);
        acc[rt][1] = wmma_bf16(a, load_frag_b(bl1, ks * 32, khalf), acc[rt][1]);
      }
    }
    __syncthreads();                      // all waves done reading buffer q&1
    if (wave == 0 && q + 2 < 4) {
      tdm_stage_2d((unsigned)(uintptr_t)(void*)&wlds[q & 1][0],
                   (const void*)((const char*)wt + (q + 2) * KQ * 2),
                   /*tile_d0_dw=*/KQ / 2, /*tile_d1=*/D, /*stride_dw=*/D / 2);
    }
  }

  // scale by edge normalization, scatter-add into destination rows
#pragma unroll
  for (int rt = 0; rt < 4; ++rt) {
#pragma unroll
    for (int i = 0; i < 8; ++i) {
      int e = base_e + rt * 16 + i + 8 * khalf;   // C/D row for VGPR i
      if (e < E) {
        float nrm = norm[(size_t)r * E + e];
        int d = dst[(size_t)r * E + e];
        atomicAdd(&agg[(size_t)d * D + nt0 * 16 + col], acc[rt][0][i] * nrm);
        atomicAdd(&agg[(size_t)d * D + (nt0 + 1) * 16 + col], acc[rt][1][i] * nrm);
      }
    }
  }
}

// ---- stage 3: fused LN -> exact GELU -> LN (rowwise) ----------------------

__global__ __launch_bounds__(256) void norm_gelu_kernel(const float* __restrict__ agg,
                                                        const float* __restrict__ osc,
                                                        const float* __restrict__ obi,
                                                        const float* __restrict__ fsc,
                                                        const float* __restrict__ fbi,
                                                        float* __restrict__ resid,
                                                        bf16* __restrict__ h2) {
  const int n = blockIdx.x;
  const int t = threadIdx.x;              // 256 threads == D
  __shared__ float s1[256], s2[256];

  float v = agg[(size_t)n * D + t];
  s1[t] = v; s2[t] = v * v;
  __syncthreads();
#pragma unroll
  for (int s = 128; s > 0; s >>= 1) {
    if (t < s) { s1[t] += s1[t + s]; s2[t] += s2[t + s]; }
    __syncthreads();
  }
  float mu  = s1[0] * (1.0f / D);
  float var = s2[0] * (1.0f / D) - mu * mu;
  float hn  = (v - mu) * rsqrtf(var + 1e-5f) * osc[t] + obi[t];
  float g   = 0.5f * hn * (1.0f + erff(hn * 0.70710678118654752f));  // exact GELU
  resid[(size_t)n * D + t] = g;

  __syncthreads();
  s1[t] = g; s2[t] = g * g;
  __syncthreads();
#pragma unroll
  for (int s = 128; s > 0; s >>= 1) {
    if (t < s) { s1[t] += s1[t + s]; s2[t] += s2[t + s]; }
    __syncthreads();
  }
  float mu2  = s1[0] * (1.0f / D);
  float var2 = s2[0] * (1.0f / D) - mu2 * mu2;
  float h    = (g - mu2) * rsqrtf(var2 + 1e-5f) * fsc[t] + fbi[t];
  h2[(size_t)n * D + t] = (bf16)h;
}

// ---- stage 4: FFN GEMM1 (D->4D) + exact GELU, bf16 out (WMMA) -------------

__global__ __launch_bounds__(256) void ffn1_kernel(const bf16* __restrict__ h2,
                                                   const bf16* __restrict__ w1,
                                                   bf16* __restrict__ g,
                                                   int N) {
  const int tile = blockIdx.x;            // 16 node rows
  const int wave = threadIdx.x >> 5;
  const int lane = threadIdx.x & 31;
  const int col  = lane & 15;
  const int khalf = lane >> 4;

  int arow_node = tile * 16 + col;
  if (arow_node >= N) arow_node = N - 1;
  const bf16* arow = h2 + (size_t)arow_node * D;

  // 64 column tiles across FF; each of 8 waves owns 8 tiles; keep 8 live
  // accumulators so every A fragment is loaded once per K step.
  f32x8 acc[8];
#pragma unroll
  for (int j = 0; j < 8; ++j) acc[j] = (f32x8){0, 0, 0, 0, 0, 0, 0, 0};

#pragma unroll
  for (int ks = 0; ks < D / 32; ++ks) {
    bf16x16 a = load_frag_a(arow, ks * 32, khalf);
#pragma unroll
    for (int j = 0; j < 8; ++j) {
      int nt = wave * 8 + j;
      const bf16* brow = w1 + (size_t)(nt * 16 + col) * D;
      acc[j] = wmma_bf16(a, load_frag_b(brow, ks * 32, khalf), acc[j]);
    }
  }

#pragma unroll
  for (int j = 0; j < 8; ++j) {
    int nt = wave * 8 + j;
#pragma unroll
    for (int i = 0; i < 8; ++i) {
      int node = tile * 16 + i + 8 * khalf;
      if (node < N) {
        float v = acc[j][i];
        v = 0.5f * v * (1.0f + erff(v * 0.70710678118654752f));
        g[(size_t)node * FF + nt * 16 + col] = (bf16)v;
      }
    }
  }
}

// ---- stage 5: FFN GEMM2 (4D->D) + residual (WMMA) -------------------------

__global__ __launch_bounds__(256) void ffn2_kernel(const bf16* __restrict__ g,
                                                   const bf16* __restrict__ w2,
                                                   const float* __restrict__ resid,
                                                   float* __restrict__ out,
                                                   int N) {
  const int tile = blockIdx.x;
  const int wave = threadIdx.x >> 5;
  const int lane = threadIdx.x & 31;
  const int col  = lane & 15;
  const int khalf = lane >> 4;

  int arow_node = tile * 16 + col;
  if (arow_node >= N) arow_node = N - 1;
  const bf16* arow = g + (size_t)arow_node * FF;

  const int nt0 = wave * 2;
  const bf16* b0row = w2 + (size_t)(nt0 * 16 + col) * FF;
  const bf16* b1row = w2 + (size_t)((nt0 + 1) * 16 + col) * FF;

  f32x8 acc0 = {0, 0, 0, 0, 0, 0, 0, 0};
  f32x8 acc1 = {0, 0, 0, 0, 0, 0, 0, 0};
#pragma unroll
  for (int ks = 0; ks < FF / 32; ++ks) {
    bf16x16 a = load_frag_a(arow, ks * 32, khalf);
    acc0 = wmma_bf16(a, load_frag_b(b0row, ks * 32, khalf), acc0);
    acc1 = wmma_bf16(a, load_frag_b(b1row, ks * 32, khalf), acc1);
  }

#pragma unroll
  for (int i = 0; i < 8; ++i) {
    int node = tile * 16 + i + 8 * khalf;
    if (node < N) {
      size_t o0 = (size_t)node * D + nt0 * 16 + col;
      size_t o1 = (size_t)node * D + (nt0 + 1) * 16 + col;
      out[o0] = resid[o0] + acc0[i];
      out[o1] = resid[o1] + acc1[i];
    }
  }
}

// ---------------------------------------------------------------------------

extern "C" void kernel_launch(void* const* d_in, const int* in_sizes, int n_in,
                              void* d_out, int out_size, void* d_ws, size_t ws_size,
                              hipStream_t stream) {
  const float* x        = (const float*)d_in[0];
  const int*   src      = (const int*)d_in[1];
  const int*   dst      = (const int*)d_in[2];
  const float* norm     = (const float*)d_in[3];
  const float* wrel_f   = (const float*)d_in[4];
  const float* wself_f  = (const float*)d_in[5];
  const float* osc      = (const float*)d_in[6];
  const float* obi      = (const float*)d_in[7];
  const float* fsc      = (const float*)d_in[8];
  const float* fbi      = (const float*)d_in[9];
  const float* w1_f     = (const float*)d_in[10];
  const float* w2_f     = (const float*)d_in[11];
  float* out = (float*)d_out;

  const int N = in_sizes[0] / D;
  const int R = in_sizes[4] / (D * D);
  const int E = in_sizes[1] / R;

  // workspace carve-up (256B aligned)
  char* ws = (char*)d_ws;
  size_t off = 0;
  auto carve = [&](size_t bytes) {
    void* p = ws + off;
    off = (off + bytes + 255) & ~(size_t)255;
    return p;
  };
  bf16*  xb     = (bf16*)carve((size_t)N * D * sizeof(bf16));
  bf16*  wrel   = (bf16*)carve((size_t)R * D * D * sizeof(bf16));
  bf16*  wself  = (bf16*)carve((size_t)D * D * sizeof(bf16));
  bf16*  w1     = (bf16*)carve((size_t)FF * D * sizeof(bf16));
  bf16*  w2     = (bf16*)carve((size_t)D * FF * sizeof(bf16));
  float* agg    = (float*)carve((size_t)N * D * sizeof(float));
  float* resid  = (float*)carve((size_t)N * D * sizeof(float));
  bf16*  h2     = (bf16*)carve((size_t)N * D * sizeof(bf16));
  bf16*  gbuf   = (bf16*)carve((size_t)N * FF * sizeof(bf16));
  (void)ws_size;

  auto blocks = [](size_t n) { return (unsigned)((n + 255) / 256); };

  // conversions (weights kept output-column-major over K; x L2-resident bf16)
  cvt_bf16_kernel<<<blocks((size_t)N * D), 256, 0, stream>>>(x, xb, (size_t)N * D);
  cvt_rel_kernel<<<blocks((size_t)R * D * D), 256, 0, stream>>>(wrel_f, wrel, R);
  cvt_bf16_kernel<<<blocks((size_t)D * D), 256, 0, stream>>>(wself_f, wself, (size_t)D * D);
  cvt_bf16_kernel<<<blocks((size_t)FF * D), 256, 0, stream>>>(w1_f, w1, (size_t)FF * D);
  cvt_bf16_kernel<<<blocks((size_t)D * FF), 256, 0, stream>>>(w2_f, w2, (size_t)D * FF);

  const unsigned ntiles   = (unsigned)((N + 15) / 16);
  const unsigned etiles64 = (unsigned)((E + 63) / 64);

  // agg = x @ Wself^T + x
  selfloop_kernel<<<ntiles, 256, 0, stream>>>(x, xb, wself, agg, N);
  // agg += scatter-add over relations of (x[src] @ W_r) * norm  (TDM-staged B)
  msg_kernel<<<dim3(etiles64, (unsigned)R), 256, 0, stream>>>(xb, wrel, src, dst, norm, agg, N, E);
  // LN -> GELU -> LN
  norm_gelu_kernel<<<(unsigned)N, 256, 0, stream>>>(agg, osc, obi, fsc, fbi, resid, h2);
  // FFN
  ffn1_kernel<<<ntiles, 256, 0, stream>>>(h2, w1, gbuf, N);
  ffn2_kernel<<<ntiles, 256, 0, stream>>>(gbuf, w2, resid, out, N);

  (void)n_in; (void)out_size;
}
